// HGATModel_9251359556147
// MI455X (gfx1250) — matrix operand order repaired and checked by reading, not compiled, not run on hardware
//
#include <hip/hip_runtime.h>
#include <hip/hip_bf16.h>

// Problem constants
#define NU 8192      // users (M)
#define NI 32768     // items (N)
#define KD 129       // feature dim
#define KP 160       // padded K: 5 chunks of 32 for bf16 WMMA
#define LSTR 168     // LDS row stride in bf16 elems (336B = 84 dwords -> conflict-free)
#define BM 64        // macro-tile M per block
#define BN 128       // macro-tile N per block

typedef __attribute__((ext_vector_type(16))) __bf16 v16bf;
typedef __attribute__((ext_vector_type(8)))  float  v8f;
typedef __attribute__((ext_vector_type(4)))  unsigned int v4u;

union Frag { v16bf f; v4u u[2]; };

// ---------------------------------------------------------------------------
// Prep: split f32 -> bf16 hi + bf16 lo (round-to-nearest-even), fold the
// signs vector (negate items' column 0), zero-pad K from 129 to 160.
// ---------------------------------------------------------------------------
__device__ __forceinline__ unsigned int f2bf_bits(float f) {
    unsigned int u = __float_as_uint(f);
    return (u + 0x7FFFu + ((u >> 16) & 1u)) >> 16;   // RNE
}

__global__ void hgat_prep(const float* __restrict__ h,
                          unsigned short* __restrict__ Ahi,
                          unsigned short* __restrict__ Alo,
                          unsigned short* __restrict__ Bhi,
                          unsigned short* __restrict__ Blo) {
    int idx = blockIdx.x * blockDim.x + threadIdx.x;
    const int TOT = (NU + NI) * KP;
    if (idx >= TOT) return;
    int row = idx / KP;
    int k   = idx - row * KP;
    float x = 0.0f;
    if (k < KD) {
        x = h[(size_t)row * KD + k];
        if (row >= NU && k == 0) x = -x;  // signs[0] = -1 applied to items
    }
    unsigned int hb = f2bf_bits(x);
    float hf = __uint_as_float(hb << 16);
    unsigned int lb = f2bf_bits(x - hf);
    if (row < NU) {
        Ahi[(size_t)row * KP + k] = (unsigned short)hb;
        Alo[(size_t)row * KP + k] = (unsigned short)lb;
    } else {
        size_t r = (size_t)(row - NU);
        Bhi[r * KP + k] = (unsigned short)hb;
        Blo[r * KP + k] = (unsigned short)lb;
    }
}

// ---------------------------------------------------------------------------
// GEMM + epilogue. Block = 256 threads = 8 wave32. Macro-tile 64x128.
// Wave w computes the 64x16 strip at column w*16: 4 accumulators of 16x16.
// Split-bf16: acc += ah*bh + al*bh + ah*bl  (3 WMMAs per k-chunk per tile).
// ---------------------------------------------------------------------------
__global__ __launch_bounds__(256) void hgat_gemm(
    const unsigned short* __restrict__ Ahi, const unsigned short* __restrict__ Alo,
    const unsigned short* __restrict__ Bhi, const unsigned short* __restrict__ Blo,
    float* __restrict__ out) {

    __shared__ __align__(16) unsigned short sA[2][BM * LSTR];   // [hi/lo]
    __shared__ __align__(16) unsigned short sB[2][BN * LSTR];

    const int tid = threadIdx.x;
    const int m0 = blockIdx.y * BM;
    const int n0 = blockIdx.x * BN;

    // Cooperative global -> LDS staging, 16B per thread per iter, coalesced.
    // A: 64 rows * 20 chunks = 1280 = 5*256 ; B: 128 * 20 = 2560 = 10*256.
    #pragma unroll
    for (int it = 0; it < 5; ++it) {
        int c = tid + it * 256;
        int row = c / 20, col = c % 20;
        *(v4u*)&sA[0][row * LSTR + col * 8] = *(const v4u*)&Ahi[(size_t)(m0 + row) * KP + col * 8];
        *(v4u*)&sA[1][row * LSTR + col * 8] = *(const v4u*)&Alo[(size_t)(m0 + row) * KP + col * 8];
    }
    #pragma unroll
    for (int it = 0; it < 10; ++it) {
        int c = tid + it * 256;
        int row = c / 20, col = c % 20;
        *(v4u*)&sB[0][row * LSTR + col * 8] = *(const v4u*)&Bhi[(size_t)(n0 + row) * KP + col * 8];
        *(v4u*)&sB[1][row * LSTR + col * 8] = *(const v4u*)&Blo[(size_t)(n0 + row) * KP + col * 8];
    }
    __syncthreads();

    const int wave = tid >> 5;
    const int lane = tid & 31;
    const int sel  = lane >> 4;   // which half-wave
    const int ln   = lane & 15;

    v8f acc0 = {0,0,0,0,0,0,0,0};
    v8f acc1 = {0,0,0,0,0,0,0,0};
    v8f acc2 = {0,0,0,0,0,0,0,0};
    v8f acc3 = {0,0,0,0,0,0,0,0};
    v8f* acc[4] = {&acc0, &acc1, &acc2, &acc3};

    const int brow = wave * 16 + ln;   // item row within B panel

    #pragma unroll
    for (int kc = 0; kc < 5; ++kc) {
        // B fragment (32x16, bf16): lane holds col=ln, K = sel?16..31:0..15,
        // i.e. 32 contiguous bytes at kc*64B + sel*32B.
        Frag bh, bl;
        {
            const unsigned short* p = &sB[0][brow * LSTR + kc * 32 + sel * 16];
            bh.u[0] = *(const v4u*)p;       bh.u[1] = *(const v4u*)(p + 8);
            const unsigned short* q = &sB[1][brow * LSTR + kc * 32 + sel * 16];
            bl.u[0] = *(const v4u*)q;       bl.u[1] = *(const v4u*)(q + 8);
        }
        #pragma unroll
        for (int tr = 0; tr < 4; ++tr) {
            // A fragment (16x32, bf16): lane holds row=ln,
            // K = sel?16..31:0..15, 32 contiguous bytes at kc*64B + sel*32B.
            const int arow = tr * 16 + ln;
            Frag ah, al;
            const unsigned short* p = &sA[0][arow * LSTR + kc * 32 + sel * 16];
            ah.u[0] = *(const v4u*)p;       ah.u[1] = *(const v4u*)(p + 8);
            const unsigned short* q = &sA[1][arow * LSTR + kc * 32 + sel * 16];
            al.u[0] = *(const v4u*)q;       al.u[1] = *(const v4u*)(q + 8);

            v8f c = *acc[tr];
            c = __builtin_amdgcn_wmma_f32_16x16x32_bf16(false, ah.f, false, bh.f, (short)0, c, false, false);
            c = __builtin_amdgcn_wmma_f32_16x16x32_bf16(false, al.f, false, bh.f, (short)0, c, false, false);
            c = __builtin_amdgcn_wmma_f32_16x16x32_bf16(false, ah.f, false, bl.f, (short)0, c, false, false);
            *acc[tr] = c;
        }
    }

    // Epilogue: -min( acosh(max(-prod, 1+eps))^2, 50 )
    // C layout: VGPR r -> M = r + 8*sel, N = ln.
    const int col = n0 + wave * 16 + ln;
    #pragma unroll
    for (int tr = 0; tr < 4; ++tr) {
        v8f c = *acc[tr];
        #pragma unroll
        for (int r = 0; r < 8; ++r) {
            int row = m0 + tr * 16 + r + sel * 8;
            float p = c[r];
            float theta = fmaxf(-p, 1.0f + 1e-7f);
            float s = __builtin_sqrtf(theta * theta - 1.0f);
            float a = __logf(theta + s);            // acosh(theta)
            float sq = fminf(a * a, 50.0f);
            out[(size_t)row * NI + col] = -sq;
        }
    }
}

// ---------------------------------------------------------------------------
extern "C" void kernel_launch(void* const* d_in, const int* in_sizes, int n_in,
                              void* d_out, int out_size, void* d_ws, size_t ws_size,
                              hipStream_t stream) {
    const float* h = (const float*)d_in[0];
    float* out = (float*)d_out;

    // ws layout (bf16/ushort): Ahi | Alo | Bhi | Blo  (~26.2 MB total)
    unsigned short* ws  = (unsigned short*)d_ws;
    unsigned short* Ahi = ws;
    unsigned short* Alo = Ahi + (size_t)NU * KP;
    unsigned short* Bhi = Alo + (size_t)NU * KP;
    unsigned short* Blo = Bhi + (size_t)NI * KP;

    const int tot = (NU + NI) * KP;
    hipLaunchKernelGGL(hgat_prep, dim3((tot + 255) / 256), dim3(256), 0, stream,
                       h, Ahi, Alo, Bhi, Blo);

    dim3 grid(NI / BN, NU / BM);   // (256, 128)
    hipLaunchKernelGGL(hgat_gemm, grid, dim3(256), 0, stream,
                       Ahi, Alo, Bhi, Blo, out);
}
